// Self_Attention_47854525612574
// MI455X (gfx1250) — matrix operand order repaired and checked by reading, not compiled, not run on hardware
//
#include <hip/hip_runtime.h>
#include <math.h>

// ---------------------------------------------------------------------------
// Types for CDNA5 WMMA (wave32): v_wmma_f32_16x16x32_bf16
// ---------------------------------------------------------------------------
typedef __bf16 bf16_t;
typedef __attribute__((ext_vector_type(8)))  __bf16 bf16x8;
typedef __attribute__((ext_vector_type(16))) __bf16 v16bf;
typedef __attribute__((ext_vector_type(8)))  float  v8f;

#define WMMA_BF16(a, b, c)                                                    \
  __builtin_amdgcn_wmma_f32_16x16x32_bf16(false, (a), false, (b), (short)0,   \
                                          (c), false, false)

// A-matrix 16x32 bf16 layout: lane L holds row M=L%16.
// lanes 0-15: K = 0..15; lanes 16-31: K = 16..31 (contiguous).
// -> two 16B chunks at (kb + 16*lhalf) and +8.
static __device__ __forceinline__ v16bf frag_ld_a(const bf16_t* p) {
  bf16x8 lo = *(const bf16x8*)p;
  bf16x8 hi = *(const bf16x8*)(p + 8);
  return __builtin_shufflevector(lo, hi, 0, 1, 2, 3, 4, 5, 6, 7, 8, 9, 10, 11,
                                 12, 13, 14, 15);
}
// B-matrix 32x16 bf16: lane L holds column N=L%16; lane half selects K 0..15
// or 16..31 (contiguous) -> two 16B chunks at (kb + 16*lhalf) and +8.
static __device__ __forceinline__ v16bf frag_ld_b(const bf16_t* p) {
  bf16x8 lo = *(const bf16x8*)p;
  bf16x8 hi = *(const bf16x8*)(p + 8);
  return __builtin_shufflevector(lo, hi, 0, 1, 2, 3, 4, 5, 6, 7, 8, 9, 10, 11,
                                 12, 13, 14, 15);
}

// ---------------------------------------------------------------------------
// fp32 -> bf16 conversion
// ---------------------------------------------------------------------------
__global__ __launch_bounds__(256) void cvt_f32_bf16(const float* __restrict__ in,
                                                    bf16_t* __restrict__ out,
                                                    int n) {
  int i = blockIdx.x * 256 + threadIdx.x;
  if (i < n) out[i] = (bf16_t)in[i];
}

// ---------------------------------------------------------------------------
// Generic WMMA GEMM: Y[m][n] = sum_k A[m][k] * W[n][k] + bias[n]
// A: M x K bf16 row-major.  W: N x K bf16 row-major (torch Linear weight).
// Block: 256 thr = 8 waves; block tile 128(M) x 64(N); wave tile 32x32.
// ---------------------------------------------------------------------------
enum { EPI_QK = 0, EPI_VT = 1, EPI_RELU = 2, EPI_RESF32 = 3 };

template <int EPI>
__global__ __launch_bounds__(256) void wmma_gemm(
    const bf16_t* __restrict__ A, const bf16_t* __restrict__ W,
    const float* __restrict__ bias, bf16_t* __restrict__ outb,
    float* __restrict__ outf, const float* __restrict__ res, int M, int N,
    int K, float scale) {
  const int lane  = threadIdx.x & 31;
  const int wave  = threadIdx.x >> 5;
  const int lhalf = lane >> 4;
  const int l15   = lane & 15;
  const int mBase = blockIdx.y * 128 + (wave >> 1) * 32;
  const int nBase = blockIdx.x * 64 + (wave & 1) * 32;
  const int koffA = lhalf * 16;
  const int koffB = lhalf * 16;

  v8f c[2][2] = {};
  for (int kb = 0; kb < K; kb += 32) {
    v16bf a[2], b[2];
#pragma unroll
    for (int i = 0; i < 2; ++i) {
      const bf16_t* ap = A + (size_t)(mBase + i * 16 + l15) * K + kb + koffA;
      __builtin_prefetch(ap + 32);  // global_prefetch_b8: next K block
      a[i] = frag_ld_a(ap);
    }
#pragma unroll
    for (int j = 0; j < 2; ++j) {
      const bf16_t* bp = W + (size_t)(nBase + j * 16 + l15) * K + kb + koffB;
      b[j] = frag_ld_b(bp);
    }
#pragma unroll
    for (int i = 0; i < 2; ++i)
#pragma unroll
      for (int j = 0; j < 2; ++j) c[i][j] = WMMA_BF16(a[i], b[j], c[i][j]);
  }

  // C/D layout: VGPR r, lane L -> M = r + 8*(L>>4), N = L&15.
#pragma unroll
  for (int i = 0; i < 2; ++i)
#pragma unroll
    for (int j = 0; j < 2; ++j)
#pragma unroll
      for (int r = 0; r < 8; ++r) {
        int m = mBase + i * 16 + r + 8 * lhalf;
        int n = nBase + j * 16 + l15;
        float v = c[i][j][r] + bias[n];
        if (EPI == EPI_QK) {
          // store to [b][h][s][64]; b=m>>10, s=m&1023, h=n>>6, d=n&63
          size_t idx = ((size_t)((m >> 10) * 16 + (n >> 6)) * 1024 +
                        (size_t)(m & 1023)) * 64 + (n & 63);
          outb[idx] = (bf16_t)(v * scale);
        } else if (EPI == EPI_VT) {
          // store transposed: [b][h][d][1024]
          size_t idx = (size_t)((m >> 10) * 16 + (n >> 6)) * 65536 +
                       (size_t)(n & 63) * 1024 + (m & 1023);
          outb[idx] = (bf16_t)v;
        } else if (EPI == EPI_RELU) {
          outb[(size_t)m * N + n] = (bf16_t)fmaxf(v, 0.0f);
        } else {  // EPI_RESF32
          size_t idx = (size_t)m * N + n;
          outf[idx] = v + res[idx];
        }
      }
}

// ---------------------------------------------------------------------------
// Attention: one block per (b,h, 16-query-row tile).  8 waves.
// Phase 1: each wave computes 8 score tiles (128 keys) with WMMA, kept in
//          registers. Softmax with key 1023 masked (-inf -> prob 0).
// Phase 2: probs -> d_out (fp32, written once) and bf16 LDS; 4 waves do
//          P x V with WMMA (V pre-transposed so B-frags are contiguous).
// ---------------------------------------------------------------------------
__global__ __launch_bounds__(256) void attention_kernel(
    const bf16_t* __restrict__ gq, const bf16_t* __restrict__ gk,
    const bf16_t* __restrict__ gvt, float* __restrict__ score_out,
    bf16_t* __restrict__ attn_out) {
  const int S = 1024, DH = 64;
  const int qblk = blockIdx.x & 63;
  const int bh   = blockIdx.x >> 6;  // b*16 + h
  const int lane  = threadIdx.x & 31;
  const int wave  = threadIdx.x >> 5;
  const int lhalf = lane >> 4;
  const int l15   = lane & 15;
  const int koffA = lhalf * 16;
  const int koffB = lhalf * 16;

  __shared__ bf16_t pLDS[16 * 1024];  // 32 KB probs
  __shared__ float  red[8][16];

  const bf16_t* qBase = gq + ((size_t)bh * S + qblk * 16) * DH;
  const bf16_t* kBase = gk + (size_t)bh * S * DH;

  // A-frags (q rows), shared by all key tiles of this wave
  v16bf aq[2];
#pragma unroll
  for (int s = 0; s < 2; ++s)
    aq[s] = frag_ld_a(qBase + (size_t)l15 * DH + s * 32 + koffA);

  // scores: 8 key tiles per wave
  v8f sc[8];
#pragma unroll
  for (int t = 0; t < 8; ++t) {
    v8f c = {};
    const int ktile = wave * 8 + t;
#pragma unroll
    for (int s = 0; s < 2; ++s) {
      v16bf bk = frag_ld_b(kBase + (size_t)(ktile * 16 + l15) * DH + s * 32 + koffB);
      c = WMMA_BF16(aq[s], bk, c);
    }
    sc[t] = c;
  }

  // ---- row max (mask global key 1023) ----
  float rowmax[8];
#pragma unroll
  for (int r = 0; r < 8; ++r) rowmax[r] = -INFINITY;
#pragma unroll
  for (int t = 0; t < 8; ++t) {
    const int col = (wave * 8 + t) * 16 + l15;
#pragma unroll
    for (int r = 0; r < 8; ++r)
      if (col != 1023) rowmax[r] = fmaxf(rowmax[r], sc[t][r]);
  }
#pragma unroll
  for (int m = 1; m <= 8; m <<= 1)
#pragma unroll
    for (int r = 0; r < 8; ++r)
      rowmax[r] = fmaxf(rowmax[r], __shfl_xor(rowmax[r], m, 32));
  if (l15 == 0) {
#pragma unroll
    for (int r = 0; r < 8; ++r) red[wave][r + 8 * lhalf] = rowmax[r];
  }
  __syncthreads();
  float rmax[8];
#pragma unroll
  for (int r = 0; r < 8; ++r) {
    float v = red[0][r + 8 * lhalf];
#pragma unroll
    for (int w = 1; w < 8; ++w) v = fmaxf(v, red[w][r + 8 * lhalf]);
    rmax[r] = v;
  }
  __syncthreads();

  // ---- exp + row sum ----
  float rowsum[8];
#pragma unroll
  for (int r = 0; r < 8; ++r) rowsum[r] = 0.0f;
#pragma unroll
  for (int t = 0; t < 8; ++t) {
    const int col = (wave * 8 + t) * 16 + l15;
#pragma unroll
    for (int r = 0; r < 8; ++r) {
      float e = (col == 1023) ? 0.0f : __expf(sc[t][r] - rmax[r]);
      sc[t][r] = e;
      rowsum[r] += e;
    }
  }
#pragma unroll
  for (int m = 1; m <= 8; m <<= 1)
#pragma unroll
    for (int r = 0; r < 8; ++r) rowsum[r] += __shfl_xor(rowsum[r], m, 32);
  if (l15 == 0) {
#pragma unroll
    for (int r = 0; r < 8; ++r) red[wave][r + 8 * lhalf] = rowsum[r];
  }
  __syncthreads();
  float rinv[8];
#pragma unroll
  for (int r = 0; r < 8; ++r) {
    float v = 0.0f;
#pragma unroll
    for (int w = 0; w < 8; ++w) v += red[w][r + 8 * lhalf];
    rinv[r] = 1.0f / v;
  }

  // ---- write probs: once to HBM (score output), once to LDS (bf16) ----
#pragma unroll
  for (int t = 0; t < 8; ++t) {
    const int col = (wave * 8 + t) * 16 + l15;
#pragma unroll
    for (int r = 0; r < 8; ++r) {
      const int row = r + 8 * lhalf;
      float p = sc[t][r] * rinv[r];
      score_out[((size_t)(bh * S + qblk * 16 + row)) * S + col] = p;
      pLDS[row * 1024 + col] = (bf16_t)p;
    }
  }
  __syncthreads();

  // ---- P x V : waves 0..3 each own one 16-wide slice of Dh=64 ----
  if (wave < 4) {
    v8f c = {};
    const bf16_t* vBase = gvt + ((size_t)bh * 64 + wave * 16 + l15) * 1024;
    for (int kb = 0; kb < S; kb += 32) {
      v16bf a  = frag_ld_a(&pLDS[l15 * 1024 + kb + koffA]);   // ds_load
      v16bf bv = frag_ld_b(vBase + kb + koffB);
      c = WMMA_BF16(a, bv, c);
    }
#pragma unroll
    for (int r = 0; r < 8; ++r) {
      size_t row = (size_t)(bh >> 4) * 1024 + qblk * 16 + r + 8 * lhalf;
      int col = (bh & 15) * 64 + wave * 16 + l15;
      attn_out[row * 1024 + col] = (bf16_t)c[r];
    }
  }
}

// ---------------------------------------------------------------------------
// LayerNorm: one block (256 thr) per row of N=1024.
// ---------------------------------------------------------------------------
__global__ __launch_bounds__(256) void layernorm_kernel(
    const float* __restrict__ in, const float* __restrict__ g,
    const float* __restrict__ be, float* __restrict__ outf,
    bf16_t* __restrict__ outb, int N) {
  __shared__ float red[256];
  const int tid = threadIdx.x;
  const size_t base = (size_t)blockIdx.x * N;

  float s = 0.0f;
  for (int c = tid; c < N; c += 256) s += in[base + c];
  red[tid] = s;
  __syncthreads();
  for (int off = 128; off > 0; off >>= 1) {
    if (tid < off) red[tid] += red[tid + off];
    __syncthreads();
  }
  const float mean = red[0] * (1.0f / N);
  __syncthreads();

  float s2 = 0.0f;
  for (int c = tid; c < N; c += 256) {
    float d = in[base + c] - mean;
    s2 += d * d;
  }
  red[tid] = s2;
  __syncthreads();
  for (int off = 128; off > 0; off >>= 1) {
    if (tid < off) red[tid] += red[tid + off];
    __syncthreads();
  }
  const float rs = rsqrtf(red[0] * (1.0f / N) + 1e-5f);

  for (int c = tid; c < N; c += 256) {
    float v = (in[base + c] - mean) * rs * g[c] + be[c];
    outf[base + c] = v;
    if (outb) outb[base + c] = (bf16_t)v;
  }
}

// ---------------------------------------------------------------------------
// Launch: full post-norm transformer layer
// ---------------------------------------------------------------------------
extern "C" void kernel_launch(void* const* d_in, const int* in_sizes, int n_in,
                              void* d_out, int out_size, void* d_ws,
                              size_t ws_size, hipStream_t stream) {
  (void)in_sizes; (void)n_in; (void)out_size; (void)ws_size;
  const int M = 8192, D = 1024, D2 = 2048;

  const float* x   = (const float*)d_in[0];
  const float* Wq  = (const float*)d_in[1];
  const float* bq  = (const float*)d_in[2];
  const float* Wk  = (const float*)d_in[3];
  const float* bk  = (const float*)d_in[4];
  const float* Wv  = (const float*)d_in[5];
  const float* bv  = (const float*)d_in[6];
  const float* Wf  = (const float*)d_in[7];
  const float* bf  = (const float*)d_in[8];
  const float* W1  = (const float*)d_in[9];
  const float* b1  = (const float*)d_in[10];
  const float* W2  = (const float*)d_in[11];
  const float* b2  = (const float*)d_in[12];
  const float* g1  = (const float*)d_in[13];
  const float* be1 = (const float*)d_in[14];
  const float* g2  = (const float*)d_in[15];
  const float* be2 = (const float*)d_in[16];

  float* out_f = (float*)d_out;                      // [8,1024,1024]
  float* score = out_f + (size_t)M * D;              // [8,16,1024,1024]

  // workspace carve-up (all sizes multiple of 256B)
  size_t off = 0;
  auto wsAlloc = [&](size_t bytes) -> char* {
    char* p = (char*)d_ws + off;
    off += (bytes + 255) & ~(size_t)255;
    return p;
  };
  bf16_t* xb   = (bf16_t*)wsAlloc((size_t)M * D * 2);   // also reused as attn
  bf16_t* wqb  = (bf16_t*)wsAlloc((size_t)D * D * 2);
  bf16_t* wkb  = (bf16_t*)wsAlloc((size_t)D * D * 2);
  bf16_t* wvb  = (bf16_t*)wsAlloc((size_t)D * D * 2);
  bf16_t* wfb  = (bf16_t*)wsAlloc((size_t)D * D * 2);
  bf16_t* w1b  = (bf16_t*)wsAlloc((size_t)D2 * D * 2);
  bf16_t* w2b  = (bf16_t*)wsAlloc((size_t)D * D2 * 2);
  bf16_t* qbuf = (bf16_t*)wsAlloc((size_t)M * D * 2);
  bf16_t* kbuf = (bf16_t*)wsAlloc((size_t)M * D * 2);
  bf16_t* vtb  = (bf16_t*)wsAlloc((size_t)M * D * 2);
  float*  prel = (float*)wsAlloc((size_t)M * D * 4);    // pre-LN1, reused pre-LN2
  float*  ln1f = (float*)wsAlloc((size_t)M * D * 4);
  bf16_t* ln1b = (bf16_t*)wsAlloc((size_t)M * D * 2);
  bf16_t* hb   = (bf16_t*)wsAlloc((size_t)M * D2 * 2);

  auto cvt = [&](const float* src, bf16_t* dst, int n) {
    cvt_f32_bf16<<<(n + 255) / 256, 256, 0, stream>>>(src, dst, n);
  };
  cvt(x, xb, M * D);
  cvt(Wq, wqb, D * D);
  cvt(Wk, wkb, D * D);
  cvt(Wv, wvb, D * D);
  cvt(Wf, wfb, D * D);
  cvt(W1, w1b, D2 * D);
  cvt(W2, w2b, D * D2);

  const dim3 blk(256);
  const dim3 gD(D / 64, M / 128);    // N=1024 GEMMs
  const dim3 gD2(D2 / 64, M / 128);  // N=2048 GEMM

  // QKV projections (q pre-scaled by 1/sqrt(64))
  wmma_gemm<EPI_QK><<<gD, blk, 0, stream>>>(xb, wqb, bq, qbuf, nullptr,
                                            nullptr, M, D, D, 0.125f);
  wmma_gemm<EPI_QK><<<gD, blk, 0, stream>>>(xb, wkb, bk, kbuf, nullptr,
                                            nullptr, M, D, D, 1.0f);
  wmma_gemm<EPI_VT><<<gD, blk, 0, stream>>>(xb, wvb, bv, vtb, nullptr,
                                            nullptr, M, D, D, 1.0f);

  // attention (xb's data is consumed; reuse as attn context buffer)
  bf16_t* attnb = xb;
  attention_kernel<<<8192, blk, 0, stream>>>(qbuf, kbuf, vtb, score, attnb);

  // out-proj + residual, LN1
  wmma_gemm<EPI_RESF32><<<gD, blk, 0, stream>>>(attnb, wfb, bf, nullptr, prel,
                                                x, M, D, D, 1.0f);
  layernorm_kernel<<<M, blk, 0, stream>>>(prel, g1, be1, ln1f, ln1b, D);

  // FFN
  wmma_gemm<EPI_RELU><<<gD2, blk, 0, stream>>>(ln1b, w1b, b1, hb, nullptr,
                                               nullptr, M, D2, D, 1.0f);
  wmma_gemm<EPI_RESF32><<<gD, blk, 0, stream>>>(hb, w2b, b2, nullptr, prel,
                                                ln1f, M, D, D2, 1.0f);
  layernorm_kernel<<<M, blk, 0, stream>>>(prel, g2, be2, out_f, nullptr, D);
}
